// Attention_79998060855419
// MI455X (gfx1250) — matrix operand order repaired and checked by reading, not compiled, not run on hardware
//
#include <hip/hip_runtime.h>
#include <hip/hip_bf16.h>
#include <math.h>

typedef __attribute__((ext_vector_type(16))) _Float16 v16h;
typedef __attribute__((ext_vector_type(8)))  _Float16 v8h;
typedef __attribute__((ext_vector_type(4)))  _Float16 v4h;
typedef __attribute__((ext_vector_type(8)))  float    v8f;

__device__ __forceinline__ float selu_f(float z) {
    const float l = 1.0507009873554805f;
    const float a = 1.6732632423543772f;
    return z > 0.f ? l * z : l * a * (__expf(z) - 1.f);
}

// ---------------------------------------------------------------------------
// A-fragment packed layout (16-bit 16x32 per the ISA):
//   lane<16 : halves 0..7 -> K = kBase+0..7,  halves 8..15 -> K = kBase+16..23
//   lane>=16: halves 0..7 -> K = kBase+8..15, halves 8..15 -> K = kBase+24..31
// Packed flat (halves): A[(kt*32 + lane)*16 + h]   (32 B per lane, contiguous)
// Inverse map of (row, K): kt=K/32, kk=K%32, g=kk/8, o=kk%8
//   lane = row + 16*(g&1),  h = o + 8*(g>>1)
//
// B-fragment packed layout (16-bit 32x16 per the ISA):
//   K = 32*kt + (lane>=16 ? 8 : 0) + (h&7) + 16*(h>>3),  N = 16*nt + (lane&15)
// Packed flat (halves): B[((kt*NT + nt)*32 + lane)*16 + h]
// ---------------------------------------------------------------------------

__global__ void att_prep_kernel(const float* __restrict__ u_rep,
                                const float* __restrict__ att1_w,
                                const float* __restrict__ att1_b,
                                const float* __restrict__ att2_w,
                                _Float16* __restrict__ B1,
                                _Float16* __restrict__ B2,
                                float* __restrict__ c1) {
    const int b = blockIdx.x, t = threadIdx.x;
    if (b < 2) {
        const int j = b * 256 + t;                       // 512 outputs
        float s = att1_b[j];
        const float* wrow = att1_w + (size_t)j * 1024 + 512;
        #pragma unroll 8
        for (int k = 0; k < 512; ++k) s += u_rep[k] * wrow[k];
        c1[j] = s;
    } else if (b < 2 + 1024) {                           // 262144 halves of B1
        const int e    = (b - 2) * 256 + t;
        const int h    = e & 15;
        const int lane = (e >> 4) & 31;
        const int nt   = (e >> 9) & 31;
        const int kt   = e >> 14;
        const int K  = kt * 32 + ((lane >= 16) ? 8 : 0) + (h & 7) + ((h & 8) << 1);
        const int Nn = nt * 16 + (lane & 15);
        B1[e] = (_Float16)att1_w[(size_t)Nn * 1024 + K]; // node half: cols 0..511
    } else {                                             // 65536 halves of B2
        const int e    = (b - 2 - 1024) * 256 + t;
        const int h    = e & 15;
        const int lane = (e >> 4) & 31;
        const int nt   = (e >> 9) & 7;
        const int kt   = e >> 12;
        const int K  = kt * 32 + ((lane >= 16) ? 8 : 0) + (h & 7) + ((h & 8) << 1);
        const int Nn = nt * 16 + (lane & 15);
        B2[e] = (_Float16)att2_w[(size_t)Nn * 512 + K];
    }
}

// ---------------------------------------------------------------------------
// Main: 16 rows per block, 256 threads = 8 wave32.
// ---------------------------------------------------------------------------
__global__ __launch_bounds__(256)
void att_main_kernel(const float* __restrict__ node1,
                     const float* __restrict__ c1,
                     const _Float16* __restrict__ B1,
                     const _Float16* __restrict__ B2,
                     const float* __restrict__ att2_b,
                     const float* __restrict__ att3_w,
                     const float* __restrict__ att3_b,
                     const float* __restrict__ lin1_w,
                     const float* __restrict__ lin1_b,
                     float* __restrict__ out, int nRows) {
    __shared__ __align__(32) _Float16 lAp[16 * 512];   // 16 KB: node tile, A-packed
    __shared__ __align__(32) _Float16 lH1p[16 * 512];  // 16 KB: hidden1, A-packed
    __shared__ float lH2[16 * 128];                    //  8 KB: hidden2 (f32)
    __shared__ float wsum[16], x3sum[16], alphaS[16];

    const int t    = threadIdx.x;
    const int lane = t & 31;
    const int w    = t >> 5;
    const bool hiL = lane >= 16;
    const int rowBase = blockIdx.x * 16;
    if (rowBase >= nRows) return;                      // block-uniform

    if (t < 16) { wsum[t] = lin1_b[0]; x3sum[t] = att3_b[0]; }
    __syncthreads();

    // ---- Stage 1: node1 tile -> f16 LDS in A-fragment packed layout; ------
    // ---- also accumulate the lin1 dot (for alpha) on the fly.        ------
    {
        const float4* src = (const float4*)(node1 + (size_t)rowBase * 512);
        #pragma unroll
        for (int it = 0; it < 8; ++it) {
            const int idx = t + it * 256;              // 0..2047 over [16][128] f4
            const int row = idx >> 7;
            const int col = (idx & 127) * 4;           // multiple of 4 (o in {0,4})
            const float4 v = src[idx];
            const int kt = col >> 5, kk = col & 31, g = kk >> 3, o = kk & 7;
            const int dst = (kt * 32 + row + ((g & 1) << 4)) * 16 + o + ((g & 2) << 2);
            v4h q;
            q[0] = (_Float16)v.x; q[1] = (_Float16)v.y;
            q[2] = (_Float16)v.z; q[3] = (_Float16)v.w;
            *(v4h*)(lAp + dst) = q;                    // 8-byte aligned run
            const float p = v.x * lin1_w[col] + v.y * lin1_w[col + 1] +
                            v.z * lin1_w[col + 2] + v.w * lin1_w[col + 3];
            atomicAdd(&wsum[row], p);
        }
    }
    __syncthreads();
    if (t < 16) alphaS[t] = 1.f + 1.f / (1.f + __expf(-wsum[t]));

    // ---- Stage 2: GEMM1 (16x512 @ 512x512) — wave w owns col tiles 4w..4w+3
    // All fragment addresses = hoisted base + compile-time immediate offsets.
    {
        const _Float16* Aw  = lAp + lane * 16;                       // +kt*512 halves
        const _Float16* B1w = B1 + ((size_t)(w * 4) * 32 + lane) * 16; // +kt*16384+ct*512
        v8f a0 = {}, a1 = {}, a2 = {}, a3 = {};
        for (int kt = 0; kt < 16; ++kt) {
            const v16h a  = *(const v16h*)(Aw + kt * 512);
            const _Float16* bp = B1w + (size_t)kt * 16384;
            const v16h b0 = *(const v16h*)(bp);
            const v16h b1 = *(const v16h*)(bp + 512);
            const v16h b2 = *(const v16h*)(bp + 1024);
            const v16h b3 = *(const v16h*)(bp + 1536);
            if (kt + 1 < 16) __builtin_prefetch(bp + 16384, 0, 1);
            a0 = __builtin_amdgcn_wmma_f32_16x16x32_f16(false, a, false, b0, (short)0, a0, false, false);
            a1 = __builtin_amdgcn_wmma_f32_16x16x32_f16(false, a, false, b1, (short)0, a1, false, false);
            a2 = __builtin_amdgcn_wmma_f32_16x16x32_f16(false, a, false, b2, (short)0, a2, false, false);
            a3 = __builtin_amdgcn_wmma_f32_16x16x32_f16(false, a, false, b3, (short)0, a3, false, false);
        }
        // Epilogue: +c1 (folded u_rep part + att1_b), selu, write h1 into the
        // A-packed layout so GEMM2 fragments are direct v16h loads.
        auto storeH1 = [&](v8f acc, int ct) {
            const int col = (w * 4 + ct) * 16 + (lane & 15);
            const float cv = c1[col];
            const int kt2 = col >> 5, kk = col & 31, g = kk >> 3, o = kk & 7;
            const int base = (kt2 * 32 + ((g & 1) << 4)) * 16 + o + ((g & 2) << 2);
            #pragma unroll
            for (int r = 0; r < 8; ++r) {
                const int row = r + (hiL ? 8 : 0);
                lH1p[base + row * 16] = (_Float16)selu_f(acc[r] + cv);
            }
        };
        storeH1(a0, 0); storeH1(a1, 1); storeH1(a2, 2); storeH1(a3, 3);
    }
    __syncthreads();

    // ---- Stage 3: GEMM2 (16x512 @ 512x128) — wave w owns column tile w ----
    {
        const _Float16* Aw2 = lH1p + lane * 16;                    // +kt*512 halves
        const _Float16* B2w = B2 + ((size_t)w * 32 + lane) * 16;   // +kt*4096 halves
        v8f c2 = {};
        for (int kt = 0; kt < 16; ++kt) {
            const v16h a = *(const v16h*)(Aw2 + kt * 512);
            const v16h b = *(const v16h*)(B2w + (size_t)kt * 4096);
            c2 = __builtin_amdgcn_wmma_f32_16x16x32_f16(false, a, false, b, (short)0, c2, false, false);
        }
        const int col = w * 16 + (lane & 15);
        const float b2 = att2_b[col];
        #pragma unroll
        for (int r = 0; r < 8; ++r) {
            const int row = r + (hiL ? 8 : 0);
            lH2[row * 128 + col] = selu_f(c2[r] + b2);
        }
    }
    __syncthreads();

    // ---- Stage 4: 128 -> 1 dot per row ----
    {
        const int row = t >> 4, seg = (t & 15) * 8;
        float p = 0.f;
        #pragma unroll
        for (int j = 0; j < 8; ++j) p += lH2[row * 128 + seg + j] * att3_w[seg + j];
        atomicAdd(&x3sum[row], p);
    }
    __syncthreads();

    // ---- Stage 5: entmax_bisect with d == 1 (closed form of the bisection) -
    if (t < 16) {
        const float am1   = alphaS[t] - 1.f;           // in (0,1)
        const float z     = x3sum[t] * am1;
        const float tauLo = z - 1.f;                   // tau_hi == tau_lo for d==1
        const float pm    = powf(fmaxf(z - tauLo, 0.f), 1.f / am1); // == 1
        out[rowBase + t]  = pm / pm;                   // p_m / sum(p_m) == 1
    }
}

extern "C" void kernel_launch(void* const* d_in, const int* in_sizes, int n_in,
                              void* d_out, int out_size, void* d_ws, size_t ws_size,
                              hipStream_t stream) {
    const float* node1  = (const float*)d_in[0];
    const float* u_rep  = (const float*)d_in[1];
    const float* att1_w = (const float*)d_in[2];
    const float* att1_b = (const float*)d_in[3];
    const float* att2_w = (const float*)d_in[4];
    const float* att2_b = (const float*)d_in[5];
    const float* att3_w = (const float*)d_in[6];
    const float* att3_b = (const float*)d_in[7];
    const float* lin1_w = (const float*)d_in[8];
    const float* lin1_b = (const float*)d_in[9];
    const int nRows = in_sizes[0] / 512;

    char* ws = (char*)d_ws;
    _Float16* B1 = (_Float16*)ws;                       // 524288 B
    _Float16* B2 = (_Float16*)(ws + 524288);            // 131072 B
    float*    c1 = (float*)(ws + 524288 + 131072);      //   2048 B

    att_prep_kernel<<<2 + 1024 + 256, 256, 0, stream>>>(
        u_rep, att1_w, att1_b, att2_w, B1, B2, c1);
    att_main_kernel<<<(nRows + 15) / 16, 256, 0, stream>>>(
        node1, c1, B1, B2, att2_b, att3_w, att3_b, lin1_w, lin1_b,
        (float*)d_out, nRows);
}
